// MultiHeadAttention_31241592111528
// MI455X (gfx1250) — matrix operand order repaired and checked by reading, not compiled, not run on hardware
//
#include <hip/hip_runtime.h>
#include <hip/hip_bf16.h>

typedef __attribute__((ext_vector_type(16))) _Float16 v16h;
typedef __attribute__((ext_vector_type(8)))  _Float16 v8h;
typedef __attribute__((ext_vector_type(4)))  _Float16 v4h;
typedef __attribute__((ext_vector_type(8)))  float    v8f;
typedef __attribute__((ext_vector_type(4)))  unsigned int u32x4;
typedef __attribute__((ext_vector_type(8)))  int      i32x8;
typedef __attribute__((ext_vector_type(4)))  int      i32x4;

#define B_   2
#define L_   2048
#define D_   1024
#define H_   16
#define HD_  64
#define BH_  (B_*H_)
#define ROWS (B_*L_)      // 4096
#define SCP  2064         // padded LDS pitch for 2048-wide score rows

// Build 16x32 f16 A-fragment from a row base pointer:
// lanes 0-15 hold K {0..15}, lanes 16-31 hold K {16..31}.
__device__ __forceinline__ v16h make_afrag(const _Float16* rowbase, int lh) {
    v8h lo = *(const v8h*)(rowbase + lh * 16);
    v8h hi = *(const v8h*)(rowbase + lh * 16 + 8);
    return __builtin_shufflevector(lo, hi, 0,1,2,3,4,5,6,7,8,9,10,11,12,13,14,15);
}

__device__ __forceinline__ v8f wmma_f16(v16h a, v16h b, v8f c) {
    return __builtin_amdgcn_wmma_f32_16x16x32_f16(false, a, false, b,
                                                  (short)0, c, false, false);
}

// ---------------------------------------------------------------------------
// Tiled GEMM: C(4096x1024) = X @ W + bias.
// Block 256 threads (8 waves), block tile 64(M) x 64(N); each wave owns a
// 16x32 slab (A-fragment reused for 2 WMMAs). Double-buffered LDS, K step 32.
//   A_IS_F32:      X is f32 (convert to f16 while staging) else f16.
//   OUT_HEAD_F16:  store f16 head-major (b*H+h, l, hd) else f32 row-major.
// ---------------------------------------------------------------------------
template<bool A_IS_F32, bool OUT_HEAD_F16>
__global__ __launch_bounds__(256) void gemm_tile(
    const void* __restrict__ Xv, const float* __restrict__ W,
    const float* __restrict__ bias, void* __restrict__ dstv)
{
    __shared__ _Float16 Xs [2][64 * 32];   // A tiles, row-major, pitch 32
    __shared__ _Float16 WsT[2][64 * 32];   // B tiles transposed [n][k], pitch 32

    const int tid  = threadIdx.x;
    const int wave = tid >> 5, lane = tid & 31;
    const int lh   = lane >> 4, lm = lane & 15;
    const int wm   = wave >> 1, wn = wave & 1;
    const int row0 = blockIdx.x * 64;
    const int col0 = blockIdx.y * 64;

    v8f acc0 = {}, acc1 = {};

    auto stage = [&](int buf, int k0) {
        {   // X tile: 64 rows x 32 k
            const int r = tid >> 2;
            const int c = (tid & 3) * 8;
            if constexpr (A_IS_F32) {
                const float* X = (const float*)Xv;
                const float4 x0 = *(const float4*)(X + (size_t)(row0 + r) * D_ + k0 + c);
                const float4 x1 = *(const float4*)(X + (size_t)(row0 + r) * D_ + k0 + c + 4);
                v8h hv;
                hv[0] = (_Float16)x0.x; hv[1] = (_Float16)x0.y;
                hv[2] = (_Float16)x0.z; hv[3] = (_Float16)x0.w;
                hv[4] = (_Float16)x1.x; hv[5] = (_Float16)x1.y;
                hv[6] = (_Float16)x1.z; hv[7] = (_Float16)x1.w;
                *(v8h*)(&Xs[buf][r * 32 + c]) = hv;
            } else {
                const _Float16* X = (const _Float16*)Xv;
                *(v8h*)(&Xs[buf][r * 32 + c]) =
                    *(const v8h*)(X + (size_t)(row0 + r) * D_ + k0 + c);
            }
        }
        {   // W tile: 32(k) x 64(n), stored transposed
            const int kr = tid >> 3;
            const int c0 = (tid & 7) * 8;
            const float4 w0 = *(const float4*)(W + (size_t)(k0 + kr) * D_ + col0 + c0);
            const float4 w1 = *(const float4*)(W + (size_t)(k0 + kr) * D_ + col0 + c0 + 4);
            _Float16* dT = &WsT[buf][0];
            dT[(c0 + 0) * 32 + kr] = (_Float16)w0.x;
            dT[(c0 + 1) * 32 + kr] = (_Float16)w0.y;
            dT[(c0 + 2) * 32 + kr] = (_Float16)w0.z;
            dT[(c0 + 3) * 32 + kr] = (_Float16)w0.w;
            dT[(c0 + 4) * 32 + kr] = (_Float16)w1.x;
            dT[(c0 + 5) * 32 + kr] = (_Float16)w1.y;
            dT[(c0 + 6) * 32 + kr] = (_Float16)w1.z;
            dT[(c0 + 7) * 32 + kr] = (_Float16)w1.w;
        }
    };

    stage(0, 0);
    for (int kt = 0; kt < D_ / 32; ++kt) {
        __syncthreads();
        if (kt + 1 < D_ / 32) stage((kt + 1) & 1, (kt + 1) * 32);   // overlap
        const _Float16* xb = &Xs [kt & 1][0];
        const _Float16* wb = &WsT[kt & 1][0];
        v16h a  = make_afrag(xb + (wm * 16 + lm) * 32, lh);
        v16h b0 = *(const v16h*)(wb + (wn * 32      + lm) * 32 + lh * 16);
        v16h b1 = *(const v16h*)(wb + (wn * 32 + 16 + lm) * 32 + lh * 16);
        acc0 = wmma_f16(a, b0, acc0);
        acc1 = wmma_f16(a, b1, acc1);
    }

    const int gc0 = col0 + wn * 32 + lm;
    const int gc1 = gc0 + 16;
    const float bv0 = bias[gc0], bv1 = bias[gc1];
#pragma unroll
    for (int r = 0; r < 8; ++r) {
        const int grow = row0 + wm * 16 + r + 8 * lh;
        if constexpr (OUT_HEAD_F16) {
            _Float16* dst = (_Float16*)dstv;
            const int bb = grow >> 11, l = grow & 2047;
            const int h0 = gc0 >> 6, hd0 = gc0 & 63;
            const int h1 = gc1 >> 6, hd1 = gc1 & 63;
            dst[(((size_t)(bb * H_ + h0) * L_) + l) * HD_ + hd0] = (_Float16)(acc0[r] + bv0);
            dst[(((size_t)(bb * H_ + h1) * L_) + l) * HD_ + hd1] = (_Float16)(acc1[r] + bv1);
        } else {
            float* out = (float*)dstv;
            out[(size_t)grow * D_ + gc0] = acc0[r] + bv0;
            out[(size_t)grow * D_ + gc1] = acc1[r] + bv1;
        }
    }
}

// ---------------------------------------------------------------------------
// Kernel 2: fused scores + softmax + PV.  block = (bh, 16 q rows), 8 waves.
// Dynamic LDS: sc[16][SCP] f32 | qt[16][64] f16 | vt[2][64][32] f16 | red[4][256] f32
// Q tile is fetched by the Tensor Data Mover (tensor_load_to_lds).
// ---------------------------------------------------------------------------
__global__ __launch_bounds__(256) void attn_fused(
    const _Float16* __restrict__ Q, const _Float16* __restrict__ K,
    const _Float16* __restrict__ V, float* __restrict__ attn,
    _Float16* __restrict__ ctx)
{
    extern __shared__ char smem_raw[];
    float*    sc  = (float*)smem_raw;              // 16*SCP floats
    _Float16* qt  = (_Float16*)(sc + 16 * SCP);    // 16*64 halves
    _Float16* vt  = qt + 16 * 64;                  // 2*64*32 halves
    float*    red = (float*)(vt + 2 * 64 * 32);    // 4*256 floats

    const int tid  = threadIdx.x;
    const int wave = tid >> 5, lane = tid & 31;
    const int lh   = lane >> 4, lm = lane & 15;
    const int bh   = blockIdx.y;
    const int q0   = blockIdx.x * 16;

    // ---- TDM: DMA the 16x64 f16 Q tile into LDS (wave 0 issues, all wait) --
    if (wave == 0) {
        const unsigned long long ga =
            (unsigned long long)(size_t)(Q + ((size_t)bh * L_ + q0) * HD_);
        const unsigned int lds_q = (unsigned int)(size_t)(void*)qt;
        u32x4 g0 = { 1u,                                    // count=1 (valid D#)
                     lds_q,                                 // lds_addr
                     (unsigned int)(ga & 0xFFFFFFFFu),      // global_addr[31:0]
                     ((unsigned int)((ga >> 32) & 0x01FFFFFFu)) | 0x80000000u }; // [56:32] | type=2
        i32x8 g1 = { 0x00010000,          // data_size=2B, no mask/flags
                     0x00400000,          // tensor_dim0 = 64 (lo16 in [31:16])
                     0x00100000,          // tensor_dim1 = 16 (lo16 in [31:16])
                     0x00400000,          // tile_dim0   = 64 ([31:16])
                     16,                  // tile_dim1   = 16
                     64,                  // tensor_dim0_stride = 64
                     0, 0 };
        i32x4 gz4 = { 0, 0, 0, 0 };
        i32x8 gz8 = { 0, 0, 0, 0, 0, 0, 0, 0 };
        __builtin_amdgcn_tensor_load_to_lds(g0, g1, gz4, gz4, gz8, 0);
        __builtin_amdgcn_s_wait_tensorcnt(0);
    }
    __syncthreads();

    // ---- Phase 1: scores = (Q Kt)/8 into LDS; wave w owns n-tiles w, w+8, ...
    for (int t = 0; t < 16; ++t) {
        const int nt = wave + t * 8;               // 0..127
        if (t + 1 < 16) {                          // prefetch next K tile
            const int ntn = wave + (t + 1) * 8;
            __builtin_prefetch(K + ((size_t)bh * L_ + ntn * 16 + lm) * HD_, 0, 1);
        }
        v8f acc = {};
#pragma unroll
        for (int kk = 0; kk < 64; kk += 32) {
            v16h a = make_afrag(qt + lm * 64 + kk, lh);
            const _Float16* kp = K + ((size_t)bh * L_ + nt * 16 + lm) * HD_ + kk + lh * 16;
            v16h b = *(const v16h*)kp;             // contiguous 32B: Kt fragment
            acc = wmma_f16(a, b, acc);
        }
#pragma unroll
        for (int r = 0; r < 8; ++r)
            sc[(r + 8 * lh) * SCP + nt * 16 + lm] = acc[r] * 0.125f;
    }
    __syncthreads();

    // ---- Phase 2: row softmax; 16 threads per row; write normalized attn
    {
        const int r = tid >> 4, j = tid & 15;
        float* row = sc + r * SCP;
        float m0 = -3.4e38f;
        for (int c = j; c < L_; c += 16) m0 = fmaxf(m0, row[c]);
        for (int off = 8; off; off >>= 1) m0 = fmaxf(m0, __shfl_xor(m0, off, 32));
        float s = 0.f;
        for (int c = j; c < L_; c += 16) { float e = __expf(row[c] - m0); row[c] = e; s += e; }
        for (int off = 8; off; off >>= 1) s += __shfl_xor(s, off, 32);
        const float inv = 1.f / s;
        float* arow = attn + ((size_t)bh * L_ + q0 + r) * L_;
        for (int c = j; c < L_; c += 16) { float p = row[c] * inv; row[c] = p; arow[c] = p; }
    }
    __syncthreads();

    // ---- Phase 3: ctx(16x64) = P @ V; waves 0-3: keys 0..1023, 4-7: 1024..2047
    const int nb = wave & 3;      // 16-col slice of Hd
    const int h2 = wave >> 2;     // key half
    v8f cacc = {};
    for (int ks = 0; ks < 32; ++ks) {
        __syncthreads();
        {   // stage V rows [h*1024+ks*32 .. +31][0..63] transposed into vt
            const int kr = tid & 31;
            const int c0 = (tid >> 5) * 8;
#pragma unroll
            for (int hh = 0; hh < 2; ++hh) {
                const int kpos = hh * 1024 + ks * 32 + kr;
                v8h vv = *(const v8h*)(V + ((size_t)bh * L_ + kpos) * HD_ + c0);
                _Float16* dT = vt + hh * (64 * 32);
#pragma unroll
                for (int c = 0; c < 8; ++c) dT[(c0 + c) * 32 + kr] = vv[c];
            }
        }
        __syncthreads();
        const float* prow = sc + lm * SCP + h2 * 1024 + ks * 32;
        v16h a;
#pragma unroll
        for (int i = 0; i < 8; ++i) {
            a[i]     = (_Float16)prow[lh * 16 + i];
            a[i + 8] = (_Float16)prow[lh * 16 + 8 + i];
        }
        v16h b = *(const v16h*)(vt + h2 * (64 * 32) + (nb * 16 + lm) * 32 + lh * 16);
        cacc = wmma_f16(a, b, cacc);
    }
    __syncthreads();
    if (h2 == 1) {
#pragma unroll
        for (int r = 0; r < 8; ++r) red[nb * 256 + (r + 8 * lh) * 16 + lm] = cacc[r];
    }
    __syncthreads();
    if (h2 == 0) {
        const int bb = bh >> 4, h = bh & 15;
#pragma unroll
        for (int r = 0; r < 8; ++r) {
            const float v = cacc[r] + red[nb * 256 + (r + 8 * lh) * 16 + lm];
            const int M = r + 8 * lh;
            ctx[((size_t)(bb * L_) + q0 + M) * D_ + h * 64 + nb * 16 + lm] = (_Float16)v;
        }
    }
}

// ---------------------------------------------------------------------------
extern "C" void kernel_launch(void* const* d_in, const int* in_sizes, int n_in,
                              void* d_out, int out_size, void* d_ws, size_t ws_size,
                              hipStream_t stream)
{
    const float* q  = (const float*)d_in[0];
    const float* k  = (const float*)d_in[1];
    const float* v  = (const float*)d_in[2];
    const float* Wq = (const float*)d_in[3];
    const float* bq = (const float*)d_in[4];
    const float* Wk = (const float*)d_in[5];
    const float* bk = (const float*)d_in[6];
    const float* Wv = (const float*)d_in[7];
    const float* bv = (const float*)d_in[8];
    const float* Wo = (const float*)d_in[9];
    const float* bo = (const float*)d_in[10];

    float* out  = (float*)d_out;                       // (B,L,D) f32
    float* attn = out + (size_t)ROWS * D_;             // (B,H,L,L) f32

    const size_t PROJ = (size_t)ROWS * D_;             // 4,194,304 halves each
    _Float16* qh = (_Float16*)d_ws;                    // head-major (BH,L,Hd)
    _Float16* kh = qh + PROJ;
    _Float16* vh = kh + PROJ;
    _Float16* ch = vh + PROJ;                          // row-major (ROWS,D)

    dim3 blk(256);
    dim3 gproj(ROWS / 64, D_ / 64);                    // 64 x 16
    gemm_tile<true, true ><<<gproj, blk, 0, stream>>>(q, Wq, bq, qh);
    gemm_tile<true, true ><<<gproj, blk, 0, stream>>>(k, Wk, bk, kh);
    gemm_tile<true, true ><<<gproj, blk, 0, stream>>>(v, Wv, bv, vh);

    dim3 gattn(L_ / 16, BH_);                          // 128 x 32
    const size_t smem = (size_t)(16 * SCP) * 4         // score rows
                      + 16 * 64 * 2                    // Q tile
                      + 2 * 64 * 32 * 2                // V chunk (transposed)
                      + 4 * 256 * 4;                   // reduction
    attn_fused<<<gattn, blk, smem, stream>>>(qh, kh, vh, attn, ch);

    gemm_tile<false, false><<<gproj, blk, 0, stream>>>(ch, Wo, bo, out);
}
